// GlobalTokenMergeModule_79070347919761
// MI455X (gfx1250) — matrix operand order repaired and checked by reading, not compiled, not run on hardware
//
#include <hip/hip_runtime.h>

#define BB    8
#define LL    512
#define DD    1024
#define NS    8192
#define GDIM  64
#define KT    256
#define RSTEP 256

typedef __attribute__((ext_vector_type(16))) _Float16 v16h;
typedef __attribute__((ext_vector_type(8)))  float    v8f;
typedef __attribute__((ext_vector_type(2)))  float    v2f;
typedef __attribute__((ext_vector_type(2)))  _Float16 v2h;

// monotone float -> uint key (ascending float order == ascending uint order)
__device__ __forceinline__ unsigned simkey(float f) {
  unsigned u = __float_as_uint(f);
  return (u & 0x80000000u) ? ~u : (u | 0x80000000u);
}

// ---------------------------------------------------------------------------
// Kernel 1: g = x @ W^T  (per-batch 512x64, K=1024) via v_wmma_f32_16x16x32_f16
// One wave per 16x16 output tile. A/B assembled per-lane in ISA VGPR layout.
// ---------------------------------------------------------------------------
__global__ void __launch_bounds__(32) proj_wmma(const float* __restrict__ x,
                                                const float* __restrict__ W,
                                                float* __restrict__ g) {
  int b = blockIdx.x, mt = blockIdx.y, nt = blockIdx.z;
  int lane = threadIdx.x;
  int half = lane >> 4, l16 = lane & 15;
  const float* xr = x + ((size_t)b * LL + (size_t)mt * 16 + l16) * DD; // A row m=l16
  const float* wr = W + ((size_t)nt * 16 + l16) * DD;                  // B col n=l16 (row of W)
  v8f acc = {};
  for (int k0 = 0; k0 < DD; k0 += 32) {
    v16h a, bm;
#pragma unroll
    for (int v = 0; v < 8; ++v) {
      // 16-bit A 16x32 layout: v<4 -> K=2v(+8*half); v>=4 -> K=16+2(v-4)(+8*half)
      int ka = (v < 4 ? 2 * v : 16 + 2 * (v - 4)) + 8 * half;
      v2f pa = *(const v2f*)(xr + k0 + ka);
      a[2 * v] = (_Float16)pa.x; a[2 * v + 1] = (_Float16)pa.y;
      // 16-bit B 32x16 layout (same K order as A): v<4 -> K=2v(+8*half); v>=4 -> K=16+2(v-4)(+8*half)
      int kb = (v < 4 ? 2 * v : 16 + 2 * (v - 4)) + 8 * half;
      v2f pb = *(const v2f*)(wr + k0 + kb);
      bm[2 * v] = (_Float16)pb.x; bm[2 * v + 1] = (_Float16)pb.y;
    }
    acc = __builtin_amdgcn_wmma_f32_16x16x32_f16(false, a, false, bm,
                                                 (short)0, acc, false, false);
  }
  // C/D f32 layout: VGPR r, lane -> (M = r + 8*half, N = l16)
  float* go = g + ((size_t)b * LL + (size_t)mt * 16) * GDIM + nt * 16;
#pragma unroll
  for (int r = 0; r < 8; ++r)
    go[(size_t)(r + 8 * half) * GDIM + l16] = acc[r];
}

// ---------------------------------------------------------------------------
// Kernel 2: gn = ||g_row||, gu = f16(g / max(gn,1e-12)). One 64-thread block
// per token.
// ---------------------------------------------------------------------------
__global__ void __launch_bounds__(64) norm_kernel(const float* __restrict__ g,
                                                  float* __restrict__ gn,
                                                  _Float16* __restrict__ gu) {
  __shared__ float red[64];
  int t = blockIdx.x;          // flat b*L + token
  int d = threadIdx.x;
  float v = g[(size_t)t * GDIM + d];
  red[d] = v * v;
  __syncthreads();
  for (int s = 32; s > 0; s >>= 1) {
    if (d < s) red[d] += red[d + s];
    __syncthreads();
  }
  float n = sqrtf(red[0]);
  if (d == 0) gn[t] = n;
  gu[(size_t)t * GDIM + d] = (_Float16)(v / fmaxf(n, 1e-12f));
}

// ---------------------------------------------------------------------------
// Kernel 3: sim = gu @ gu^T per batch (512x512, K=64): 2 WMMAs per 16x16 tile.
// ---------------------------------------------------------------------------
__global__ void __launch_bounds__(32) sim_wmma(const _Float16* __restrict__ gu,
                                               float* __restrict__ sim) {
  int b = blockIdx.x, mt = blockIdx.y, nt = blockIdx.z;
  int lane = threadIdx.x, half = lane >> 4, l16 = lane & 15;
  const _Float16* ar = gu + ((size_t)b * LL + (size_t)mt * 16 + l16) * GDIM;
  const _Float16* br = gu + ((size_t)b * LL + (size_t)nt * 16 + l16) * GDIM;
  v8f acc = {};
#pragma unroll
  for (int k0 = 0; k0 < GDIM; k0 += 32) {
    v16h a, bm;
#pragma unroll
    for (int v = 0; v < 8; ++v) {
      int ka = (v < 4 ? 2 * v : 16 + 2 * (v - 4)) + 8 * half;
      v2h pa = *(const v2h*)(ar + k0 + ka);
      a[2 * v] = pa.x; a[2 * v + 1] = pa.y;
      int kb = (v < 4 ? 2 * v : 16 + 2 * (v - 4)) + 8 * half;
      v2h pb = *(const v2h*)(br + k0 + kb);
      bm[2 * v] = pb.x; bm[2 * v + 1] = pb.y;
    }
    acc = __builtin_amdgcn_wmma_f32_16x16x32_f16(false, a, false, bm,
                                                 (short)0, acc, false, false);
  }
  float* so = sim + ((size_t)b * LL + (size_t)mt * 16) * LL + nt * 16;
#pragma unroll
  for (int r = 0; r < 8; ++r)
    so[(size_t)(r + 8 * half) * LL + l16] = acc[r];
}

// ---------------------------------------------------------------------------
// Kernel 4: greedy disjoint pair selection == 256 rounds of argmax over the
// still-available upper-triangle, lazy per-row best cache. One 512-thread
// workgroup per batch; thread tid owns row tid.
// Ordering key matches lexsort((pj, pi, -sim)): max sim, then min i, then min j.
// ---------------------------------------------------------------------------
__global__ void __launch_bounds__(LL) greedy_kernel(const float* __restrict__ sim,
                                                    int* __restrict__ keep_idx,
                                                    int* __restrict__ part_idx) {
  __shared__ int used[LL];
  __shared__ int partner[LL];
  __shared__ int isj[LL];
  __shared__ unsigned long long rowbest[LL];  // (simkey<<9)|(511-j); 0 == none
  __shared__ unsigned long long red[LL];
  int b = blockIdx.x;
  int tid = threadIdx.x;
  const float* sb = sim + (size_t)b * LL * LL;
  used[tid] = 0; partner[tid] = -1; isj[tid] = 0;
  __syncthreads();

  // initial per-row best (upper triangle only: j > i)
  {
    unsigned long long best = 0;
    const float* srow = sb + (size_t)tid * LL;
    for (int j = tid + 1; j < LL; ++j) {
      unsigned long long k =
          ((unsigned long long)simkey(srow[j]) << 9) | (unsigned long long)(511 - j);
      if (k > best) best = k;
    }
    rowbest[tid] = best;
  }
  __syncthreads();

  for (int round = 0; round < RSTEP; ++round) {
    // validate / lazily recompute this row's cached best
    unsigned long long rb = rowbest[tid];
    if (used[tid]) {
      rowbest[tid] = 0;
    } else if (rb != 0) {
      int bj = 511 - (int)(rb & 0x1FFull);
      if (used[bj]) {
        unsigned long long best = 0;
        const float* srow = sb + (size_t)tid * LL;
        for (int j = tid + 1; j < LL; ++j) {
          if (used[j]) continue;
          unsigned long long k =
              ((unsigned long long)simkey(srow[j]) << 9) | (unsigned long long)(511 - j);
          if (k > best) best = k;
        }
        rowbest[tid] = best;
      }
    }
    __syncthreads();

    // full key: (simkey<<18) | ((511-i)<<9) | (511-j)  -> max == greedy pick
    unsigned long long fk = 0;
    rb = rowbest[tid];
    if (!used[tid] && rb != 0) {
      unsigned long long sk = rb >> 9;
      unsigned long long jj = rb & 0x1FFull;
      fk = (sk << 18) | ((unsigned long long)(511 - tid) << 9) | jj;
    }
    red[tid] = fk;
    __syncthreads();
    for (int s = LL / 2; s > 0; s >>= 1) {
      if (tid < s) { if (red[tid + s] > red[tid]) red[tid] = red[tid + s]; }
      __syncthreads();
    }
    if (tid == 0) {
      unsigned long long sel = red[0];
      int i = 511 - (int)((sel >> 9) & 0x1FFull);
      int j = 511 - (int)(sel & 0x1FFull);
      partner[i] = j;
      isj[j] = 1;
      used[i] = 1; used[j] = 1;
    }
    __syncthreads();
  }

  // kept tokens in original order (exactly KT of them on a complete graph)
  if (tid == 0) {
    int k = 0;
    for (int t = 0; t < LL; ++t)
      if (!isj[t]) { keep_idx[b * KT + k] = t; part_idx[b * KT + k] = partner[t]; ++k; }
  }
}

// ---------------------------------------------------------------------------
// Kernel 5: xm = has_p ? (wi*x[t] + wj*x[p]) / (wi+wj+1e-8) : x[t]
// ---------------------------------------------------------------------------
__global__ void __launch_bounds__(256) merge_x(const float* __restrict__ x,
                                               const float* __restrict__ gn,
                                               const int* __restrict__ keep_idx,
                                               const int* __restrict__ part_idx,
                                               float* __restrict__ xm) {
  int row = blockIdx.x;           // b*KT + k
  int b = row / KT;
  int t = keep_idx[row], p = part_idx[row];
  int c = threadIdx.x * 4;
  float4 vt = *(const float4*)(x + ((size_t)b * LL + t) * DD + c);
  float4 o;
  if (p >= 0) {
    float wi = gn[b * LL + t], wj = gn[b * LL + p];
    float inv = 1.0f / (wi + wj + 1e-8f);
    float4 vp = *(const float4*)(x + ((size_t)b * LL + p) * DD + c);
    o.x = (wi * vt.x + wj * vp.x) * inv;
    o.y = (wi * vt.y + wj * vp.y) * inv;
    o.z = (wi * vt.z + wj * vp.z) * inv;
    o.w = (wi * vt.w + wj * vp.w) * inv;
  } else {
    o = vt;
  }
  *(float4*)(xm + (size_t)row * DD + c) = o;
}

// ---------------------------------------------------------------------------
// Kernel 6: sm = has_p ? s[t] + s[p] : s[t]   (bandwidth-dominant: ~200 MB)
// ---------------------------------------------------------------------------
__global__ void __launch_bounds__(256) merge_s(const float* __restrict__ s,
                                               const int* __restrict__ keep_idx,
                                               const int* __restrict__ part_idx,
                                               float* __restrict__ sm) {
  int row = blockIdx.x;           // b*KT + k
  int b = row / KT;
  int t = keep_idx[row], p = part_idx[row];
  int c = blockIdx.y * 1024 + threadIdx.x * 4;
  float4 vt = *(const float4*)(s + ((size_t)b * LL + t) * NS + c);
  if (p >= 0) {
    float4 vp = *(const float4*)(s + ((size_t)b * LL + p) * NS + c);
    vt.x += vp.x; vt.y += vp.y; vt.z += vp.z; vt.w += vp.w;
  }
  *(float4*)(sm + (size_t)row * NS + c) = vt;
}

extern "C" void kernel_launch(void* const* d_in, const int* in_sizes, int n_in,
                              void* d_out, int out_size, void* d_ws, size_t ws_size,
                              hipStream_t stream) {
  const float* x   = (const float*)d_in[0];   // (B,L,D)
  const float* src = (const float*)d_in[1];   // (B,L,N)
  const float* W   = (const float*)d_in[2];   // (GD,D)
  // d_in[3] = target_len scalar; shapes are compile-time constants here.

  char* w = (char*)d_ws;
  float*    g    = (float*)(w);                       // 8*512*64  f32 = 1,048,576 B
  _Float16* gu   = (_Float16*)(w + 1048576);          // 8*512*64  f16 =   524,288 B
  float*    gn   = (float*)(w + 1572864);             // 8*512     f32 =    16,384 B
  float*    sim  = (float*)(w + 1589248);             // 8*512*512 f32 = 8,388,608 B
  int*      keep = (int*)(w + 9977856);               // 8*256     i32 =     8,192 B
  int*      part = (int*)(w + 9986048);               // 8*256     i32 =     8,192 B

  float* xm = (float*)d_out;                          // (B,KT,D)
  float* sm = xm + (size_t)BB * KT * DD;              // (B,KT,NS)

  proj_wmma<<<dim3(BB, LL / 16, GDIM / 16), 32, 0, stream>>>(x, W, g);
  norm_kernel<<<BB * LL, 64, 0, stream>>>(g, gn, gu);
  sim_wmma<<<dim3(BB, LL / 16, LL / 16), 32, 0, stream>>>(gu, sim);
  greedy_kernel<<<BB, LL, 0, stream>>>(sim, keep, part);
  merge_x<<<BB * KT, DD / 4, 0, stream>>>(x, gn, keep, part, xm);
  merge_s<<<dim3(BB * KT, NS / 1024), 256, 0, stream>>>(src, keep, part, sm);
}
